// loTeNet_76785425318286
// MI455X (gfx1250) — matrix operand order repaired and checked by reading, not compiled, not run on hardware
//
#include <hip/hip_runtime.h>

// ---------------------------------------------------------------------------
// LoTeNet on MI455X (gfx1250): MPS levels as skinny GEMMs on bf16 WMMA with
// bf16x3 (hi/lo split) emulation of fp32, f32 accumulation. Memory-bound by
// the cores tensors (~430 MB fp32 streamed once), so bf16x3 compute is free.
//
// Per patch p (one workgroup, 8 waves = 4 batch-tiles x 2 col-tiles of the
// 64x32 result):
//   site 0 : v = phi(64xC2) @ cores[p,0,:,0,:]           (row-0 only!)
//   site s : v = (phi (x) v)(64 x C2*32) @ cores[p,s]    (contiguous (C2*32)x32)
//   label  : out = v(64x32) @ label[p,:,:,0]^T
// Mode-specialized straight-line inner loops; v fragment hoisted to registers
// per site; cores staged double-buffered (stage c+1 while computing c).
// ---------------------------------------------------------------------------

typedef __attribute__((ext_vector_type(16))) __bf16    v16bf;
typedef __attribute__((ext_vector_type(8)))  float     v8f;
typedef __attribute__((ext_vector_type(8)))  unsigned  v8u;
typedef __attribute__((ext_vector_type(4)))  unsigned  v4u;
typedef __attribute__((ext_vector_type(4)))  float     v4f;

#define CHUNK 8            // k-steps (K=32 each) staged in LDS per chunk
#define SB_CSTRIDE 36      // u32 per column in s_B: 16 hi + 16 lo + 4 pad
#define SB_KSTRIDE (32 * SB_CSTRIDE)   // 1152 u32 per k-step
#define SB_WORDS (CHUNK * SB_KSTRIDE)  // u32 per staging buffer
#define SV_STRIDE 36       // padded f32 row stride of the v buffers

union Frag {               // trivial union: ext-vector members only
  v16bf bf;
  v8u   u;
  v4u   q[2];
};

// pack hi bf16 halves of (a0,a1) into one u32: a0.hi16 -> [15:0], a1.hi16 -> [31:16]
__device__ __forceinline__ unsigned hi_pair(float a0, float a1) {
  return __builtin_amdgcn_perm(__float_as_uint(a1), __float_as_uint(a0), 0x07060302u);
}
__device__ __forceinline__ unsigned lo_pair(float a0, float a1) {
  float l0 = a0 - __uint_as_float(__float_as_uint(a0) & 0xFFFF0000u);
  float l1 = a1 - __uint_as_float(__float_as_uint(a1) & 0xFFFF0000u);
  return hi_pair(l0, l1);
}

__device__ __forceinline__ v8f wmma_bf16(Frag a, Frag b, v8f c) {
  // (neg_a, A, neg_b, B, c_mod, C, reuse_a, reuse_b)
  return __builtin_amdgcn_wmma_f32_16x16x32_bf16(false, a.bf, false, b.bf,
                                                 (short)0, c, false, false);
}

// split 16 fp32 A values (fragment element order) into hi/lo bf16 fragments
__device__ __forceinline__ void build_A(const float* av, Frag& ah, Frag& al) {
#pragma unroll
  for (int qq = 0; qq < 8; ++qq) {
    ah.u[qq] = hi_pair(av[2 * qq], av[2 * qq + 1]);
    al.u[qq] = lo_pair(av[2 * qq], av[2 * qq + 1]);
  }
}

// load one B hi/lo fragment pair from fragment-ordered LDS (4x ds_load_b128)
__device__ __forceinline__ void load_B(const unsigned* bp, Frag& bh, Frag& bl) {
  bh.q[0] = *(const v4u*)(bp);
  bh.q[1] = *(const v4u*)(bp + 8);
  bl.q[0] = *(const v4u*)(bp + 16);
  bl.q[1] = *(const v4u*)(bp + 24);
}

// v-buffer permutation: A-fragment k-order [0-7,16-23,8-15,24-31] (swap bits 3,4)
__device__ __forceinline__ int vperm(int k) {
  return (k & 7) | ((k & 8) << 1) | ((k & 16) >> 1);
}

// stage one CHUNK of contiguous cores rows (CHUNK*32 x 32 f32) into
// fragment-ordered hi/lo bf16 planes
__device__ __forceinline__ void stage_b_chunk(unsigned* sb, const float* src, int tid) {
  for (int e = tid; e < CHUNK * 16 * 8; e += 256) {
    int kp = e >> 3;                   // k-pair within chunk
    int cg = e & 7;                    // column group of 4
    v4f r0 = *(const v4f*)(src + (size_t)(2 * kp) * 32 + cg * 4);
    v4f r1 = *(const v4f*)(src + (size_t)(2 * kp) * 32 + 32 + cg * 4);
    int base = (kp >> 4) * SB_KSTRIDE + (kp & 15);
#pragma unroll
    for (int t = 0; t < 4; ++t) {
      int cc = (cg * 4 + t) * SB_CSTRIDE;
      sb[base + cc]      = hi_pair(r0[t], r1[t]);
      sb[base + cc + 16] = lo_pair(r0[t], r1[t]);
    }
  }
}

// ---------------------------------------------------------------------------
// Kernel 0: untangle the unfold/reshape permutation of x (64,3,128,128) into
// x1 layout [S=4][P=256][C=48][b=64] so level-1 phi staging is coalesced.
// ---------------------------------------------------------------------------
__global__ __launch_bounds__(256) void gather_x_kernel(const float* __restrict__ x,
                                                       float* __restrict__ x1) {
  int idx = blockIdx.x * blockDim.x + threadIdx.x;     // 64*3*128*128 elements
  float val = x[idx];
  int b = idx / 49152;
  int r = idx - b * 49152;               // c*16384 + h*128 + w
  int c = r >> 14;  int hw = r & 16383;
  int h = hw >> 7;  int w  = hw & 127;
  int hb = h >> 5, hi = h & 31, wb = w >> 5, wj = w & 31;
  int flat2 = c * 16384 + hb * 4096 + wb * 1024 + hi * 32 + wj;
  int a1 = flat2 / 1536; int rem = flat2 - a1 * 1536;
  int a2 = rem / 48;     int a3  = rem - a2 * 48;
  int jb = a2 >> 4, ji = a2 & 15;
  int pb = a3 / 16, pi = a3 - pb * 16;
  int q = a1 * 6 + jb * 3 + pb;
  int C = q >> 2, S = q & 3, P = ji * 16 + pi;
  x1[(((size_t)(S * 256 + P)) * 48 + C) * 64 + b] = val;
}

// ---------------------------------------------------------------------------
// Per-channel BatchNorm stats over contiguous [channel][P*64] block; emits
// scale/shift fused into the next level's phi staging.
// ---------------------------------------------------------------------------
__global__ __launch_bounds__(256) void bn_stats_kernel(const float* __restrict__ y,
                                                       const float* __restrict__ g,
                                                       const float* __restrict__ bt,
                                                       float* __restrict__ sc_sh,
                                                       int P) {
  int o = blockIdx.x;
  int N = P * 64;
  const float* ch = y + (size_t)o * N;
  float s = 0.f, q = 0.f;
  for (int i = threadIdx.x; i < N; i += 256) { float v = ch[i]; s += v; q += v * v; }
  __shared__ float rs[256], rq[256];
  rs[threadIdx.x] = s; rq[threadIdx.x] = q;
  __syncthreads();
  for (int st = 128; st > 0; st >>= 1) {
    if (threadIdx.x < st) { rs[threadIdx.x] += rs[threadIdx.x + st];
                            rq[threadIdx.x] += rq[threadIdx.x + st]; }
    __syncthreads();
  }
  if (threadIdx.x == 0) {
    float mean = rs[0] / N;
    float var  = rq[0] / N - mean * mean;
    float scale = g[o] * rsqrtf(var + 1e-5f);
    sc_sh[o]      = scale;
    sc_sh[32 + o] = bt[o] - mean * scale;
  }
}

// ---------------------------------------------------------------------------
// MPS level kernel (levels 1-3).
// ---------------------------------------------------------------------------
__global__ __launch_bounds__(256) void mps_level_kernel(
    const float* __restrict__ phi_src,  // L1: x1 [s][p][C][b]; L2/3: y_prev [o][pp][b]
    const float* __restrict__ cores,    // (P,S,C2,32,32)
    const float* __restrict__ label,    // (P,32,32,32)
    const float* __restrict__ bn,       // scale[32], shift[32] (null for level 1)
    float* __restrict__ y_out,          // [o][p][b]
    int level, int C, int S, int P, int Pprev) {
  const int C2 = 2 * C;
  __shared__ float    s_phi[96 * 64];          // full 2C features, [f][b]
  __shared__ float    s_v[2][64 * SV_STRIDE];  // v chain, A-fragment k-order
  __shared__ unsigned s_B[2][SB_WORDS];        // double-buffered hi/lo bf16 planes

  const int p    = blockIdx.x;
  const int tid  = threadIdx.x;
  const int wave = tid >> 5;
  const int lane = tid & 31;
  const int hf   = lane >> 4;          // wave half (0/1)
  const int lb   = lane & 15;          // lane-in-half
  const int btile = wave >> 1, ntile = wave & 1;
  const int bA   = btile * 16 + lb;    // batch row this lane owns in A / C
  const int colB = ntile * 16 + lb;    // output column this lane owns in B / C
  const int bOff = colB * SB_CSTRIDE + hf * 4;   // lane's base into a B k-step

  int cur = 0;
  for (int s = 0; s <= S; ++s) {       // s == S is the label stage
    // ---- stage full phi (x and 1-x) for this site: s_phi[f][b], f in [0,C2) ----
    if (s < S) {
      if (level == 1) {
        const float* src = phi_src + ((size_t)(s * P + p) * C) * 64;
        for (int e = tid; e < C * 64; e += 256) {
          float v = src[e];
          s_phi[e]          = v;
          s_phi[C * 64 + e] = 1.0f - v;
        }
      } else {
        int pp;
        if (level == 2) pp = (((s >> 1) * 8) + (p >> 3)) * 16 + (s & 1) * 8 + (p & 7);
        else            pp = (((s >> 1) * 4) + (p >> 2)) * 8  + (s & 1) * 4 + (p & 3);
        for (int e = tid; e < C * 64; e += 256) {
          int o = e >> 6, b = e & 63;
          float v = phi_src[((size_t)(o * Pprev + pp)) * 64 + b] * bn[o] + bn[32 + o];
          s_phi[e]          = v;
          s_phi[C * 64 + e] = 1.0f - v;
        }
      }
    }
    __syncthreads();

    v8f acc = {0.f, 0.f, 0.f, 0.f, 0.f, 0.f, 0.f, 0.f};

    if (s == 0) {
      // ================= site 0: A = phi, B = cores[p,0,:,0,:] =================
      const int ksteps = C2 >> 5;                    // 3 or 2, fits one buffer
      const float* src = cores + (size_t)(p * S) * C2 * 1024;
      for (int e = tid; e < ksteps * 16 * 32; e += 256) {
        int kp = e >> 5, col = e & 31;
        float a0 = src[(size_t)(2 * kp) * 1024 + col];
        float a1 = src[(size_t)(2 * kp) * 1024 + 1024 + col];
        int base = (kp >> 4) * SB_KSTRIDE + (kp & 15) + col * SB_CSTRIDE;
        s_B[0][base]      = hi_pair(a0, a1);
        s_B[0][base + 16] = lo_pair(a0, a1);
      }
      __syncthreads();
      for (int kk = 0; kk < ksteps; ++kk) {
        float av[16];
#pragma unroll
        for (int j = 0; j < 16; ++j) {
          int f = kk * 32 + j + (j & 8) + hf * 8;    // A k-map
          av[j] = s_phi[f * 64 + bA];
        }
        Frag ah, al; build_A(av, ah, al);
        Frag bh, bl; load_B(&s_B[0][kk * SB_KSTRIDE + bOff], bh, bl);
        acc = wmma_bf16(ah, bh, acc);
        acc = wmma_bf16(ah, bl, acc);
        acc = wmma_bf16(al, bh, acc);
      }
      __syncthreads();
    } else if (s < S) {
      // ======== chained site: A = phi (x) v, B = contiguous cores block ========
      // v fragment is invariant across the site: hoist into registers.
      const float* vp = &s_v[cur][bA * SV_STRIDE + hf * 16];
      v4f w0 = *(const v4f*)(vp);
      v4f w1 = *(const v4f*)(vp + 4);
      v4f w2 = *(const v4f*)(vp + 8);
      v4f w3 = *(const v4f*)(vp + 12);
      const float* csrc = cores + (size_t)(p * S + s) * C2 * 1024;
      const int nchunks = C2 / CHUNK;                // 12 or 8 (C2 % CHUNK == 0)

      stage_b_chunk(s_B[0], csrc, tid);
      __syncthreads();
      for (int c = 0; c < nchunks; ++c) {
        if (c + 1 < nchunks) {                       // overlap: stage c+1 now
          stage_b_chunk(s_B[(c + 1) & 1], csrc + (size_t)(c + 1) * CHUNK * 1024, tid);
          if (c + 2 < nchunks)
            __builtin_prefetch((const char*)(csrc + (size_t)(c + 2) * CHUNK * 1024) +
                               tid * 128, 0, 3);
        }
        const unsigned* sb = s_B[c & 1];
        float phv[CHUNK];
#pragma unroll
        for (int t = 0; t < CHUNK; ++t) phv[t] = s_phi[(c * CHUNK + t) * 64 + bA];
#pragma unroll
        for (int kkl = 0; kkl < CHUNK; ++kkl) {
          float ph = phv[kkl];
          float av[16];
#pragma unroll
          for (int t = 0; t < 4; ++t) {
            av[t]      = w0[t] * ph;
            av[4 + t]  = w1[t] * ph;
            av[8 + t]  = w2[t] * ph;
            av[12 + t] = w3[t] * ph;
          }
          Frag ah, al; build_A(av, ah, al);
          Frag bh, bl; load_B(&sb[kkl * SB_KSTRIDE + bOff], bh, bl);
          acc = wmma_bf16(ah, bh, acc);
          acc = wmma_bf16(ah, bl, acc);
          acc = wmma_bf16(al, bh, acc);
        }
        __syncthreads();                             // one barrier per chunk
      }
    } else {
      // ================= label stage: A = v, B[j][o] = label[p,o,j,0] ==========
      const float* vp = &s_v[cur][bA * SV_STRIDE + hf * 16];
      v4f w0 = *(const v4f*)(vp);
      v4f w1 = *(const v4f*)(vp + 4);
      v4f w2 = *(const v4f*)(vp + 8);
      v4f w3 = *(const v4f*)(vp + 12);
      for (int e = tid; e < 512; e += 256) {
        int kp = e >> 5, o = e & 31;
        float a0 = label[(size_t)((p * 32 + o) * 32 + 2 * kp) * 32];
        float a1 = label[(size_t)((p * 32 + o) * 32 + 2 * kp + 1) * 32];
        int base = kp + o * SB_CSTRIDE;
        s_B[0][base]      = hi_pair(a0, a1);
        s_B[0][base + 16] = lo_pair(a0, a1);
      }
      __syncthreads();
      float av[16];
#pragma unroll
      for (int t = 0; t < 4; ++t) {
        av[t] = w0[t]; av[4 + t] = w1[t]; av[8 + t] = w2[t]; av[12 + t] = w3[t];
      }
      Frag ah, al; build_A(av, ah, al);
      Frag bh, bl; load_B(&s_B[0][bOff], bh, bl);
      acc = wmma_bf16(ah, bh, acc);
      acc = wmma_bf16(ah, bl, acc);
      acc = wmma_bf16(al, bh, acc);
      __syncthreads();
    }

    // ---- write v_next in A-fragment k-order (C/D layout: VGPR r -> M=r+hf*8) ----
    const int nxt = cur ^ 1;
    const int pn  = vperm(ntile * 16 + lb);
#pragma unroll
    for (int r = 0; r < 8; ++r)
      s_v[nxt][(btile * 16 + r + hf * 8) * SV_STRIDE + pn] = acc[r];
    __syncthreads();
    cur = nxt;
  }

  // ---- coalesced store of out[b][o] -> y_out[o][p][b] ----
  for (int e = tid; e < 64 * 32; e += 256) {
    int o = e >> 6, b = e & 63;
    y_out[((size_t)o * P + p) * 64 + b] = s_v[cur][b * SV_STRIDE + vperm(o)];
  }
}

// ---------------------------------------------------------------------------
// Final MPS: single module, S=16 sites, C=32 (C2=64), P=1. Same machinery.
// ---------------------------------------------------------------------------
__global__ __launch_bounds__(256) void mps_final_kernel(
    const float* __restrict__ y3,      // [o][16][b], pre-BN
    const float* __restrict__ bn,      // scale[32], shift[32]
    const float* __restrict__ coresF,  // (16,64,32,32)
    const float* __restrict__ labelF,  // (1,32,32)
    float* __restrict__ out) {
  const int C = 32, C2 = 64, S = 16;
  __shared__ float    s_phi[64 * 64];
  __shared__ float    s_v[2][64 * SV_STRIDE];
  __shared__ unsigned s_B[2][SB_WORDS];

  const int tid  = threadIdx.x;
  const int wave = tid >> 5, lane = tid & 31;
  const int hf = lane >> 4, lb = lane & 15;
  const int btile = wave >> 1, ntile = wave & 1;
  const int bA = btile * 16 + lb, colB = ntile * 16 + lb;
  const int bOff = colB * SB_CSTRIDE + hf * 4;

  int cur = 0;
  for (int s = 0; s < S; ++s) {
    for (int e = tid; e < C * 64; e += 256) {
      int o = e >> 6, b = e & 63;
      float v = y3[((size_t)(o * 16 + s)) * 64 + b] * bn[o] + bn[32 + o];
      s_phi[e]          = v;
      s_phi[C * 64 + e] = 1.0f - v;
    }
    __syncthreads();

    v8f acc = {0.f, 0.f, 0.f, 0.f, 0.f, 0.f, 0.f, 0.f};

    if (s == 0) {
      const int ksteps = C2 >> 5;                    // 2
      for (int e = tid; e < ksteps * 16 * 32; e += 256) {
        int kp = e >> 5, col = e & 31;
        float a0 = coresF[(size_t)(2 * kp) * 1024 + col];
        float a1 = coresF[(size_t)(2 * kp) * 1024 + 1024 + col];
        int base = (kp >> 4) * SB_KSTRIDE + (kp & 15) + col * SB_CSTRIDE;
        s_B[0][base]      = hi_pair(a0, a1);
        s_B[0][base + 16] = lo_pair(a0, a1);
      }
      __syncthreads();
      for (int kk = 0; kk < ksteps; ++kk) {
        float av[16];
#pragma unroll
        for (int j = 0; j < 16; ++j) {
          int f = kk * 32 + j + (j & 8) + hf * 8;
          av[j] = s_phi[f * 64 + bA];
        }
        Frag ah, al; build_A(av, ah, al);
        Frag bh, bl; load_B(&s_B[0][kk * SB_KSTRIDE + bOff], bh, bl);
        acc = wmma_bf16(ah, bh, acc);
        acc = wmma_bf16(ah, bl, acc);
        acc = wmma_bf16(al, bh, acc);
      }
      __syncthreads();
    } else {
      const float* vp = &s_v[cur][bA * SV_STRIDE + hf * 16];
      v4f w0 = *(const v4f*)(vp);
      v4f w1 = *(const v4f*)(vp + 4);
      v4f w2 = *(const v4f*)(vp + 8);
      v4f w3 = *(const v4f*)(vp + 12);
      const float* csrc = coresF + (size_t)s * C2 * 1024;
      const int nchunks = C2 / CHUNK;                // 8

      stage_b_chunk(s_B[0], csrc, tid);
      __syncthreads();
      for (int c = 0; c < nchunks; ++c) {
        if (c + 1 < nchunks) {
          stage_b_chunk(s_B[(c + 1) & 1], csrc + (size_t)(c + 1) * CHUNK * 1024, tid);
          if (c + 2 < nchunks)
            __builtin_prefetch((const char*)(csrc + (size_t)(c + 2) * CHUNK * 1024) +
                               tid * 128, 0, 3);
        }
        const unsigned* sb = s_B[c & 1];
        float phv[CHUNK];
#pragma unroll
        for (int t = 0; t < CHUNK; ++t) phv[t] = s_phi[(c * CHUNK + t) * 64 + bA];
#pragma unroll
        for (int kkl = 0; kkl < CHUNK; ++kkl) {
          float ph = phv[kkl];
          float av[16];
#pragma unroll
          for (int t = 0; t < 4; ++t) {
            av[t]      = w0[t] * ph;
            av[4 + t]  = w1[t] * ph;
            av[8 + t]  = w2[t] * ph;
            av[12 + t] = w3[t] * ph;
          }
          Frag ah, al; build_A(av, ah, al);
          Frag bh, bl; load_B(&sb[kkl * SB_KSTRIDE + bOff], bh, bl);
          acc = wmma_bf16(ah, bh, acc);
          acc = wmma_bf16(ah, bl, acc);
          acc = wmma_bf16(al, bh, acc);
        }
        __syncthreads();
      }
    }

    const int nxt = cur ^ 1;
    const int pn  = vperm(ntile * 16 + lb);
#pragma unroll
    for (int r = 0; r < 8; ++r)
      s_v[nxt][(btile * 16 + r + hf * 8) * SV_STRIDE + pn] = acc[r];
    __syncthreads();
    cur = nxt;
  }

  // out[b] = sum_j v[b,j] * labelF[0,j,0]
  if (tid < 64) {
    float a = 0.f;
#pragma unroll
    for (int j = 0; j < 32; ++j)
      a += s_v[cur][tid * SV_STRIDE + vperm(j)] * labelF[(size_t)j * 32];
    out[tid] = a;
  }
}

// ---------------------------------------------------------------------------
extern "C" void kernel_launch(void* const* d_in, const int* in_sizes, int n_in,
                              void* d_out, int out_size, void* d_ws, size_t ws_size,
                              hipStream_t stream) {
  (void)in_sizes; (void)n_in; (void)out_size; (void)ws_size;
  const float* x      = (const float*)d_in[0];
  const float* cores1 = (const float*)d_in[1];
  const float* label1 = (const float*)d_in[2];
  const float* g1     = (const float*)d_in[3];
  const float* b1     = (const float*)d_in[4];
  const float* cores2 = (const float*)d_in[5];
  const float* label2 = (const float*)d_in[6];
  const float* g2     = (const float*)d_in[7];
  const float* b2     = (const float*)d_in[8];
  const float* cores3 = (const float*)d_in[9];
  const float* label3 = (const float*)d_in[10];
  const float* g3     = (const float*)d_in[11];
  const float* b3     = (const float*)d_in[12];
  const float* coresF = (const float*)d_in[13];
  const float* labelF = (const float*)d_in[14];

  float* ws  = (float*)d_ws;
  float* x1  = ws;               // 64*49152      = 3,145,728 f
  float* y1  = x1 + 3145728;     // 32*256*64     =   524,288 f
  float* y2  = y1 + 524288;      // 32*64*64      =   131,072 f
  float* y3  = y2 + 131072;      // 32*16*64      =    32,768 f
  float* bn1 = y3 + 32768;       // 64 f
  float* bn2 = bn1 + 64;         // 64 f
  float* bn3 = bn2 + 64;         // 64 f

  gather_x_kernel<<<12288, 256, 0, stream>>>(x, x1);
  mps_level_kernel<<<256, 256, 0, stream>>>(x1, cores1, label1, nullptr, y1,
                                            1, 48, 4, 256, 0);
  bn_stats_kernel<<<32, 256, 0, stream>>>(y1, g1, b1, bn1, 256);
  mps_level_kernel<<<64, 256, 0, stream>>>(y1, cores2, label2, bn1, y2,
                                           2, 32, 4, 64, 256);
  bn_stats_kernel<<<32, 256, 0, stream>>>(y2, g2, b2, bn2, 64);
  mps_level_kernel<<<16, 256, 0, stream>>>(y2, cores3, label3, bn2, y3,
                                           3, 32, 4, 16, 64);
  bn_stats_kernel<<<32, 256, 0, stream>>>(y3, g3, b3, bn3, 16);
  mps_final_kernel<<<1, 256, 0, stream>>>(y3, bn3, coresF, labelF, (float*)d_out);
}
